// S4Layer_85298050498930
// MI455X (gfx1250) — matrix operand order repaired and checked by reading, not compiled, not run on hardware
//
#include <hip/hip_runtime.h>
#include <hip/hip_bf16.h>

// ---------------- types ----------------
typedef __attribute__((ext_vector_type(16))) _Float16 v16h;
typedef __attribute__((ext_vector_type(8)))  float    v8f;
typedef __attribute__((ext_vector_type(4)))  float    v4f;
typedef __attribute__((ext_vector_type(4)))  unsigned int v4u;

#define BATCH   8
#define SEQ     4096
#define DM      128     // d_model
#define DS      128     // d_state
#define KR_LEN  4160    // 4096 + 64 pad (reversed kernel, zero tail)
#define XT_STRIDE 4128  // padded row stride for transposed x

// ---------------- kernel 1: W = C @ B (128x128) ----------------
__global__ void k_w(const float* __restrict__ C, const float* __restrict__ B,
                    float* __restrict__ W) {
    int idx = blockIdx.x * blockDim.x + threadIdx.x;   // 16384 threads
    int m = idx >> 7, j = idx & 127;
    float s = 0.f;
#pragma unroll 4
    for (int t = 0; t < DS; ++t) s += C[m * DS + t] * B[t * DM + j];
    W[idx] = s;
}

// ---------------- kernel 2: Psum[j,l] = sum_m exp(dt[m]*Lam[j]*l) ----------------
__global__ void k_psum(const float* __restrict__ Lam, const float* __restrict__ log_dt,
                       float* __restrict__ Psum) {
    __shared__ float rate[DM];
    int j = blockIdx.y;
    int l = blockIdx.x * 256 + threadIdx.x;
    if (threadIdx.x < DM)
        rate[threadIdx.x] = __expf(log_dt[threadIdx.x]) * Lam[j];
    __syncthreads();
    float fl = (float)l;
    float s = 0.f;
#pragma unroll 4
    for (int m = 0; m < DM; ++m) s += __expf(rate[m] * fl);
    Psum[j * SEQ + l] = s;
}

// ---------------- kernel 3: KR[m][t] = K[m][4096-t] (reversed, zero padded) ----------------
__global__ void k_kr(const float* __restrict__ W, const float* __restrict__ Psum,
                     float* __restrict__ KRf) {
    int m = blockIdx.y;
    int t = blockIdx.x * 256 + threadIdx.x;
    if (t >= KR_LEN) return;
    float v = 0.f;
    if (t >= 1 && t <= SEQ) {
        int l = SEQ - t;
        const float* w = W + m * DS;
        float s = 0.f;
#pragma unroll 4
        for (int j = 0; j < DS; ++j) s += w[j] * Psum[j * SEQ + l];
        v = s;
    }
    KRf[m * KR_LEN + t] = v;
}

// ---------------- kernel 4: xT[m][b][l] = (f16) x[b][l][m] ----------------
__global__ void k_xt(const float* __restrict__ x, _Float16* __restrict__ xT) {
    __shared__ float tile[16][17];
    int b = blockIdx.z, m0 = blockIdx.y * 16, l0 = blockIdx.x * 16;
    int tx = threadIdx.x, ty = threadIdx.y;
    tile[ty][tx] = x[((size_t)b * SEQ + (l0 + ty)) * DM + m0 + tx];
    __syncthreads();
    xT[((size_t)((m0 + ty) * BATCH + b)) * XT_STRIDE + l0 + tx] = (_Float16)tile[tx][ty];
}

// ---------------- fragment loader for the conv GEMM ----------------
// A (16x32 f16, Toeplitz of reversed kernel):
//   lane<16 : row M=lane,     halves 0..7 -> K=cb+0..7, halves 8..15 -> K=cb+16..23, cb=0
//   lane>=16: row M=lane-16,  same with cb=8
// B (32x16 f16): lane<16 col=lane holds K=0..7,16..23; lane>=16 col=lane-16 holds K=8..15,24..31
__device__ __forceinline__ void load_frags(const float* __restrict__ kr,
                                           const _Float16* __restrict__ xr,
                                           int tb0, int cb, int kb, int k0,
                                           v16h& a, v16h& b) {
    int tb = tb0 + k0;                           // 4-byte aligned f32 addresses
    v4f q0 = *(const v4f*)(kr + tb);
    v4f q1 = *(const v4f*)(kr + tb + 4);
    v4f q2 = *(const v4f*)(kr + tb + 16);
    v4f q3 = *(const v4f*)(kr + tb + 20);
#pragma unroll
    for (int h = 0; h < 4; ++h) {
        a[h]      = (_Float16)q0[h];
        a[4 + h]  = (_Float16)q1[h];
        a[8 + h]  = (_Float16)q2[h];
        a[12 + h] = (_Float16)q3[h];
    }
    union { v16h v; v4u q[2]; } bu;
    const _Float16* bp = xr + k0 + kb;           // 16-byte aligned
    bu.q[0] = *(const v4u*)(bp);
    bu.q[1] = *(const v4u*)(bp + 16);
    b = bu.v;
}

// ---------------- kernel 5: causal conv via WMMA + skip ----------------
__global__ void __launch_bounds__(256) k_conv(const float* __restrict__ KRf,
                                              const _Float16* __restrict__ xT,
                                              const float* __restrict__ x,
                                              const float* __restrict__ Dv,
                                              float* __restrict__ out) {
    int wid  = blockIdx.x * 8 + (threadIdx.x >> 5);   // global wave id
    int lane = threadIdx.x & 31;
    int i = wid >> 7;            // output row-tile 0..255 (uniform per block)
    int m = wid & 127;           // channel
    int li = i << 4;

    int r  = lane & 15;
    int cb = (lane >> 4) << 3;   // A half-K offset: 0 or 8
    int kb = (lane >> 4) << 3;   // B half-K offset: 0 or 8
    int n  = lane & 15;          // output column (batch), only n<8 valid

    const float*    kr = KRf + m * KR_LEN;
    const _Float16* xr = xT + (size_t)(m * BATCH + (n & 7)) * XT_STRIDE;
    int tb0 = SEQ - li - r + cb;                  // reversed-kernel base index

    v8f acc0 = {}; v8f acc1 = {};
    int nk = (li + 47) >> 5;                      // ceil((li+16)/32) k-tiles

    int j0 = 0;
    for (; j0 + 2 <= nk; j0 += 2) {
        v16h a0, b0, a1, b1;
        load_frags(kr, xr, tb0, cb, kb, (j0 << 5), a0, b0);
        load_frags(kr, xr, tb0, cb, kb, ((j0 + 1) << 5), a1, b1);
        acc0 = __builtin_amdgcn_wmma_f32_16x16x32_f16(false, a0, false, b0,
                                                      (short)0, acc0, false, false);
        acc1 = __builtin_amdgcn_wmma_f32_16x16x32_f16(false, a1, false, b1,
                                                      (short)0, acc1, false, false);
    }
    if (j0 < nk) {
        v16h a0, b0;
        load_frags(kr, xr, tb0, cb, kb, (j0 << 5), a0, b0);
        acc0 = __builtin_amdgcn_wmma_f32_16x16x32_f16(false, a0, false, b0,
                                                      (short)0, acc0, false, false);
    }

    if (n < 8) {                                  // columns 8..15 are padding
        int rowAdd = (lane >> 4) << 3;            // M = p (+8 for lanes>=16)
        float dv = Dv[m];
#pragma unroll
        for (int p = 0; p < 8; ++p) {
            int l = li + p + rowAdd;
            size_t idx = ((size_t)n * SEQ + l) * DM + m;
            out[idx] = (acc0[p] + acc1[p]) + dv * x[idx];
        }
    }
}

// ---------------- launcher ----------------
extern "C" void kernel_launch(void* const* d_in, const int* in_sizes, int n_in,
                              void* d_out, int out_size, void* d_ws, size_t ws_size,
                              hipStream_t stream) {
    const float* x      = (const float*)d_in[0];  // (8,4096,128)
    const float* Lam    = (const float*)d_in[1];  // (128,)
    const float* B      = (const float*)d_in[2];  // (128,128)
    const float* C      = (const float*)d_in[3];  // (128,128)
    const float* Dv     = (const float*)d_in[4];  // (128,)
    const float* log_dt = (const float*)d_in[5];  // (128,)
    float* out = (float*)d_out;

    char* ws = (char*)d_ws;
    size_t off = 0;
    float* W    = (float*)(ws + off); off += (size_t)DM * DS * 4;              // 64 KB
    float* Psum = (float*)(ws + off); off += (size_t)DS * SEQ * 4;             // 2 MB
    float* KRf  = (float*)(ws + off); off += (size_t)DM * KR_LEN * 4;          // 2.03 MB
    _Float16* xT = (_Float16*)(ws + off);                                      // 8.06 MB

    // 1) W = C @ B
    k_w<<<dim3(64), dim3(256), 0, stream>>>(C, B, W);
    // 2) Psum[s,l]
    k_psum<<<dim3(SEQ / 256, DS), dim3(256), 0, stream>>>(Lam, log_dt, Psum);
    // 3) reversed padded kernel KR = reverse(W @ Psum)
    k_kr<<<dim3((KR_LEN + 255) / 256, DM), dim3(256), 0, stream>>>(W, Psum, KRf);
    // 4) transpose x -> f16 xT[m][b][l]
    k_xt<<<dim3(SEQ / 16, DM / 16, BATCH), dim3(16, 16), 0, stream>>>(x, xT);
    // 5) WMMA Toeplitz conv + skip: 32768 waves, 8 per block (uniform work/block)
    k_conv<<<dim3(32768 / 8), dim3(256), 0, stream>>>(KRf, xT, x, Dv, out);
}
